// PointNetPlusPlus_15101105012950
// MI455X (gfx1250) — matrix operand order repaired and checked by reading, not compiled, not run on hardware
//
#include <hip/hip_runtime.h>

typedef __attribute__((ext_vector_type(16))) _Float16 v16h;
typedef __attribute__((ext_vector_type(8)))  _Float16 v8h;
typedef __attribute__((ext_vector_type(8)))  float    v8f;

// ---------------------------------------------------------------------------
// WMMA GEMM:  C[M,N] = act(A[M,K] @ W[K,N] + bias), W pre-packed TRANSPOSED
// as Wt[N][K] (f16, zero padded). K multiple of 32 (compile-time), N multiple
// of 16, M multiple of 128. One wave -> one 16x16 C tile, 8 waves / block.
// Fragments are loaded directly from global as contiguous b128s:
//   A frag lane(l): m = l&15, chunks A[m][k0+8*(l>>4)] and A[m][k0+16+8*(l>>4)]
//   B frag lane(l): n = l&15, chunks Wt[n][k0+8*(l>>4)] and Wt[n][k0+16+8*(l>>4)]
// ---------------------------------------------------------------------------
template <int KT, int RELU, int OF16>
__global__ __launch_bounds__(256) void wmma_gemm_kernel(
    const _Float16* __restrict__ A, const _Float16* __restrict__ Wt,
    const float* __restrict__ bias, void* __restrict__ out,
    int M, int N, int Nstore, int ldc)
{
    const int tid  = threadIdx.x;
    const int w    = tid >> 5;
    const int lane = tid & 31;
    const int mt   = blockIdx.y * 8 + w;
    const int nt   = blockIdx.x;
    const int m0   = mt * 16, n0 = nt * 16;
    const int nn   = lane & 15;
    const int half = lane >> 4;

    const _Float16* arow = A  + (size_t)(m0 + nn) * KT + half * 8;
    const _Float16* wrow = Wt + (size_t)(n0 + nn) * KT + half * 8;

    v8f c;
    {
        float bv = bias[n0 + nn];
        #pragma unroll
        for (int r = 0; r < 8; ++r) c[r] = bv;
    }

    #pragma unroll
    for (int k0 = 0; k0 < KT; k0 += 32) {
        v8h a0 = *(const v8h*)(arow + k0);
        v8h a1 = *(const v8h*)(arow + k0 + 16);
        v8h b0 = *(const v8h*)(wrow + k0);
        v8h b1 = *(const v8h*)(wrow + k0 + 16);
        v16h af = __builtin_shufflevector(a0, a1, 0, 1, 2, 3, 4, 5, 6, 7,
                                                  8, 9, 10, 11, 12, 13, 14, 15);
        v16h bf = __builtin_shufflevector(b0, b1, 0, 1, 2, 3, 4, 5, 6, 7,
                                                  8, 9, 10, 11, 12, 13, 14, 15);
        c = __builtin_amdgcn_wmma_f32_16x16x32_f16(
                false, af, false, bf, (short)0, c, false, false);
    }

    const int n = n0 + nn;
    if (n < Nstore) {
        #pragma unroll
        for (int r = 0; r < 8; ++r) {
            int m = m0 + half * 8 + r;
            float v = c[r];
            if (RELU) v = fmaxf(v, 0.f);
            if (OF16) ((_Float16*)out)[(size_t)m * ldc + n] = (_Float16)v;
            else      ((float*)out)[(size_t)m * ldc + n]    = v;
        }
    }
}

// ---------------------------------------------------------------------------
// Farthest point sampling (single block, sequential rounds, mind[] in LDS)
// ---------------------------------------------------------------------------
__global__ __launch_bounds__(1024) void fps_kernel(
    const float* __restrict__ pos, int n, int m, int* __restrict__ out)
{
    extern __shared__ float mind[];           // n floats (dynamic LDS)
    __shared__ float rv[1024];
    __shared__ int   ri[1024];
    __shared__ int   curs;
    const int t = threadIdx.x;

    for (int j = t; j < n; j += 1024) mind[j] = 1e10f;
    if (t == 0) { curs = 0; out[0] = 0; }
    __syncthreads();

    int cur = 0;
    for (int i = 1; i < m; ++i) {
        float cx = pos[cur * 3 + 0], cy = pos[cur * 3 + 1], cz = pos[cur * 3 + 2];
        float bestv = -1.f; int besti = 0;
        for (int j = t; j < n; j += 1024) {
            float dx = pos[j * 3 + 0] - cx;
            float dy = pos[j * 3 + 1] - cy;
            float dz = pos[j * 3 + 2] - cz;
            float d  = dx * dx + dy * dy + dz * dz;
            float md = fminf(mind[j], d);
            mind[j]  = md;
            if (md > bestv) { bestv = md; besti = j; }
        }
        rv[t] = bestv; ri[t] = besti;
        __syncthreads();
        for (int s = 512; s > 0; s >>= 1) {
            if (t < s) {
                float ov = rv[t + s]; int oi = ri[t + s];
                if (ov > rv[t] || (ov == rv[t] && oi < ri[t])) { rv[t] = ov; ri[t] = oi; }
            }
            __syncthreads();
        }
        if (t == 0) { curs = ri[0]; out[i] = ri[0]; }
        __syncthreads();
        cur = curs;
    }
}

// ---------------------------------------------------------------------------
// kNN (one thread per query; sorted insertion list)
// ---------------------------------------------------------------------------
template <int KNB>
__global__ void knn_kernel(const float* __restrict__ src, int n,
                           const float* __restrict__ dst, int m,
                           int* __restrict__ out)
{
    int i = blockIdx.x * blockDim.x + threadIdx.x;
    if (i >= m) return;
    float px = dst[i * 3 + 0], py = dst[i * 3 + 1], pz = dst[i * 3 + 2];
    float bd[KNB]; int bi[KNB];
    for (int q = 0; q < KNB; ++q) { bd[q] = 3e38f; bi[q] = 0; }
    for (int j = 0; j < n; ++j) {
        __builtin_prefetch(src + 3 * (j + 512), 0, 0);
        float dx = src[j * 3 + 0] - px;
        float dy = src[j * 3 + 1] - py;
        float dz = src[j * 3 + 2] - pz;
        float d  = dx * dx + dy * dy + dz * dz;
        if (d < bd[KNB - 1]) {
            int p = KNB - 1;
            while (p > 0 && bd[p - 1] > d) {
                bd[p] = bd[p - 1]; bi[p] = bi[p - 1]; --p;
            }
            bd[p] = d; bi[p] = j;
        }
    }
    for (int q = 0; q < KNB; ++q) out[(size_t)i * KNB + q] = bi[q];
}

// 3-NN with inverse-distance weights (for feature propagation)
__global__ void knn3w_kernel(const float* __restrict__ src, int n,
                             const float* __restrict__ dst, int m,
                             int* __restrict__ idx3, float* __restrict__ w3)
{
    int i = blockIdx.x * blockDim.x + threadIdx.x;
    if (i >= m) return;
    float px = dst[i * 3 + 0], py = dst[i * 3 + 1], pz = dst[i * 3 + 2];
    float d0 = 3e38f, d1 = 3e38f, d2 = 3e38f;
    int   i0 = 0, i1 = 0, i2 = 0;
    for (int j = 0; j < n; ++j) {
        float dx = src[j * 3 + 0] - px;
        float dy = src[j * 3 + 1] - py;
        float dz = src[j * 3 + 2] - pz;
        float d  = dx * dx + dy * dy + dz * dz;
        if (d < d0)      { d2 = d1; i2 = i1; d1 = d0; i1 = i0; d0 = d; i0 = j; }
        else if (d < d1) { d2 = d1; i2 = i1; d1 = d;  i1 = j; }
        else if (d < d2) { d2 = d;  i2 = j; }
    }
    float w0 = 1.f / (sqrtf(fmaxf(d0, 0.f)) + 1e-8f);
    float w1 = 1.f / (sqrtf(fmaxf(d1, 0.f)) + 1e-8f);
    float w2 = 1.f / (sqrtf(fmaxf(d2, 0.f)) + 1e-8f);
    float s  = w0 + w1 + w2;
    idx3[i * 3 + 0] = i0; idx3[i * 3 + 1] = i1; idx3[i * 3 + 2] = i2;
    w3[i * 3 + 0] = w0 / s; w3[i * 3 + 1] = w1 / s; w3[i * 3 + 2] = w2 / s;
}

__global__ void interp_kernel(const float* __restrict__ xs, int C,
                              const int* __restrict__ idx3, const float* __restrict__ w3,
                              float* __restrict__ out, int total)
{
    int t = blockIdx.x * blockDim.x + threadIdx.x;
    if (t >= total) return;
    int i = t / C, c = t % C;
    out[t] = w3[i * 3 + 0] * xs[(size_t)idx3[i * 3 + 0] * C + c]
           + w3[i * 3 + 1] * xs[(size_t)idx3[i * 3 + 1] * C + c]
           + w3[i * 3 + 2] * xs[(size_t)idx3[i * 3 + 2] * C + c];
}

// ---------------------------------------------------------------------------
// Input builders / pooling / packing
// ---------------------------------------------------------------------------
__global__ void gather3_kernel(const float* __restrict__ src, const int* __restrict__ idx,
                               float* __restrict__ dst, int m)
{
    int i = blockIdx.x * blockDim.x + threadIdx.x;
    if (i >= m) return;
    int s = idx[i];
    dst[i * 3 + 0] = src[s * 3 + 0];
    dst[i * 3 + 1] = src[s * 3 + 1];
    dst[i * 3 + 2] = src[s * 3 + 2];
}

__global__ void build_sa1_kernel(const float* __restrict__ x, const float* __restrict__ pos,
                                 const float* __restrict__ pos1, const int* __restrict__ nbr,
                                 _Float16* __restrict__ A, int rows)
{
    int r = blockIdx.x * blockDim.x + threadIdx.x;
    if (r >= rows) return;
    int i = r >> 5;                       // 32 neighbors
    int s = nbr[r];
    _Float16* a = A + (size_t)r * 32;
    a[0] = (_Float16)x[s * 4 + 0];
    a[1] = (_Float16)x[s * 4 + 1];
    a[2] = (_Float16)x[s * 4 + 2];
    a[3] = (_Float16)x[s * 4 + 3];
    a[4] = (_Float16)(pos[s * 3 + 0] - pos1[i * 3 + 0]);
    a[5] = (_Float16)(pos[s * 3 + 1] - pos1[i * 3 + 1]);
    a[6] = (_Float16)(pos[s * 3 + 2] - pos1[i * 3 + 2]);
    for (int q = 7; q < 32; ++q) a[q] = (_Float16)0.f;
}

__global__ void build_sa2_kernel(const float* __restrict__ x1, const float* __restrict__ pos1,
                                 const float* __restrict__ pos2, const int* __restrict__ nbr,
                                 _Float16* __restrict__ A, int rows)
{
    int r = blockIdx.x * blockDim.x + threadIdx.x;
    if (r >= rows) return;
    int i = r >> 6;                       // 64 neighbors
    int s = nbr[r];
    _Float16* a = A + (size_t)r * 160;
    for (int c = 0; c < 128; ++c) a[c] = (_Float16)x1[(size_t)s * 128 + c];
    a[128] = (_Float16)(pos1[s * 3 + 0] - pos2[i * 3 + 0]);
    a[129] = (_Float16)(pos1[s * 3 + 1] - pos2[i * 3 + 1]);
    a[130] = (_Float16)(pos1[s * 3 + 2] - pos2[i * 3 + 2]);
    for (int q = 131; q < 160; ++q) a[q] = (_Float16)0.f;
}

__global__ void build_fp_kernel(const float* __restrict__ xc, int Cc,
                                const float* __restrict__ xi, int Ci,
                                _Float16* __restrict__ A, int Kpad, int m)
{
    int r = blockIdx.x * blockDim.x + threadIdx.x;
    if (r >= m) return;
    _Float16* a = A + (size_t)r * Kpad;
    for (int c = 0; c < Cc; ++c) a[c] = (_Float16)xc[(size_t)r * Cc + c];
    for (int c = 0; c < Ci; ++c) a[Cc + c] = (_Float16)xi[(size_t)r * Ci + c];
    for (int c = Cc + Ci; c < Kpad; ++c) a[c] = (_Float16)0.f;
}

__global__ void maxpool_kernel(const _Float16* __restrict__ in, _Float16* __restrict__ out,
                               int Nd, int k, int C)
{
    int t = blockIdx.x * blockDim.x + threadIdx.x;
    if (t >= Nd * C) return;
    int i = t / C, c = t % C;
    float mv = -3e38f;
    for (int j = 0; j < k; ++j)
        mv = fmaxf(mv, (float)in[((size_t)i * k + j) * C + c]);
    out[t] = (_Float16)mv;
}

// pack weights TRANSPOSED: wt[n][k] = w[k][n], zero padded to [Npad][Kpad]
__global__ void packwt_kernel(const float* __restrict__ w, _Float16* __restrict__ wt,
                              int K, int N, int Kpad, int Npad)
{
    int t = blockIdx.x * blockDim.x + threadIdx.x;
    if (t >= Kpad * Npad) return;
    int nn = t / Kpad, kk = t % Kpad;
    wt[t] = (kk < K && nn < N) ? (_Float16)w[(size_t)kk * N + nn] : (_Float16)0.f;
}

__global__ void packb_kernel(const float* __restrict__ b, float* __restrict__ bp,
                             int N, int Npad)
{
    int t = blockIdx.x * blockDim.x + threadIdx.x;
    if (t >= Npad) return;
    bp[t] = (t < N) ? b[t] : 0.f;
}

// ---------------------------------------------------------------------------
// Host orchestration
// ---------------------------------------------------------------------------
template <int KT, int RELU, int OF16>
static void lg(const _Float16* A, const _Float16* Wt, const float* b, void* out,
               int M, int N, int Nstore, int ldc, hipStream_t s)
{
    dim3 grid(N / 16, M / 128);
    wmma_gemm_kernel<KT, RELU, OF16><<<grid, 256, 0, s>>>(A, Wt, b, out, M, N, Nstore, ldc);
}

extern "C" void kernel_launch(void* const* d_in, const int* in_sizes, int n_in,
                              void* d_out, int out_size, void* d_ws, size_t ws_size,
                              hipStream_t stream)
{
    (void)n_in; (void)out_size; (void)ws_size;

    // ---- locate inputs (dict-flattening order is ambiguous; detect by sizes)
    static const int perm_sorted[28] = {15,14,17,16,11,10,13,12,23,22,25,24,
                                        19,18,21,20,5,4,7,6,1,0,3,2,27,26,9,8};
    const float* x;  const float* pos;  const float* P[28];
    if (in_sizes[0] == 65536) {
        x   = (const float*)d_in[0];
        pos = (const float*)d_in[1];
        if (in_sizes[3] == 448) {                       // insertion order
            for (int i = 0; i < 28; ++i) P[i] = (const float*)d_in[3 + i];
        } else {                                        // params sorted alphabetically
            for (int i = 0; i < 28; ++i) P[i] = (const float*)d_in[3 + perm_sorted[i]];
        }
    } else {                                            // fully sorted top-level keys
        x   = (const float*)d_in[30];
        pos = (const float*)d_in[29];
        for (int i = 0; i < 28; ++i) P[i] = (const float*)d_in[1 + perm_sorted[i]];
    }

    // ---- workspace bump allocator
    char* ws = (char*)d_ws;
    size_t off = 0;
    auto alloc = [&](size_t bytes) -> char* {
        off = (off + 255) & ~(size_t)255;
        char* p = ws + off;
        off += bytes;
        return p;
    };

    int*       idx1  = (int*)alloc(8192 * 4);
    float*     pos1  = (float*)alloc(8192 * 3 * 4);
    int*       nbr1  = (int*)alloc((size_t)8192 * 32 * 4);
    _Float16*  P1h   = (_Float16*)alloc((size_t)8192 * 64 * 2);
    _Float16*  G1h   = (_Float16*)alloc((size_t)8192 * 128 * 2);
    float*     x1f   = (float*)alloc((size_t)8192 * 128 * 4);
    int*       idx2  = (int*)alloc(2048 * 4);
    float*     pos2  = (float*)alloc(2048 * 3 * 4);
    int*       nbr2  = (int*)alloc((size_t)2048 * 64 * 4);
    _Float16*  P2h   = (_Float16*)alloc((size_t)2048 * 256 * 2);
    _Float16*  G2h   = (_Float16*)alloc((size_t)2048 * 256 * 2);
    float*     x2f   = (float*)alloc((size_t)2048 * 256 * 4);
    int*       idxA  = (int*)alloc(2048 * 3 * 4);
    float*     wA    = (float*)alloc(2048 * 3 * 4);
    float*     xint2 = (float*)alloc((size_t)2048 * 128 * 4);
    _Float16*  Afp2  = (_Float16*)alloc((size_t)2048 * 384 * 2);
    _Float16*  T1h   = (_Float16*)alloc((size_t)2048 * 256 * 2);
    float*     xf2f  = (float*)alloc((size_t)2048 * 256 * 4);
    int*       idxB  = (int*)alloc(2048 * 3 * 4);
    float*     wB    = (float*)alloc(2048 * 3 * 4);
    float*     xint1 = (float*)alloc((size_t)2048 * 4 * 4);
    _Float16*  Afp1  = (_Float16*)alloc((size_t)2048 * 288 * 2);
    _Float16*  Tb1h  = (_Float16*)alloc((size_t)2048 * 256 * 2);
    _Float16*  xf1h  = (_Float16*)alloc((size_t)2048 * 256 * 2);

    // packed (transposed, padded) weights/biases
    static const struct { int wi, bi, K, N, Kpad, Npad; } L[14] = {
        {0,1,7,64,32,64},       {2,3,64,64,64,64},      {4,5,64,128,64,128},
        {6,7,128,128,128,128},  {8,9,131,128,160,128},  {10,11,128,256,128,256},
        {12,13,256,256,256,256},{14,15,256,256,256,256},{16,17,384,256,384,256},
        {18,19,256,256,256,256},{20,21,260,256,288,256},{22,23,256,256,256,256},
        {24,25,256,3,256,16},   {26,27,256,64,256,64},
    };
    _Float16* wp[14]; float* bp[14];
    for (int l = 0; l < 14; ++l) {
        wp[l] = (_Float16*)alloc((size_t)L[l].Kpad * L[l].Npad * 2);
        bp[l] = (float*)alloc((size_t)L[l].Npad * 4);
    }

    // big arena with stage-based reuse (136 MiB)
    char* arena = alloc(142606336ULL);
    _Float16* A1  = (_Float16*)(arena);                    // [262144,32]
    _Float16* H1  = (_Float16*)(arena + (16ULL << 20));    // [262144,64]
    _Float16* H2  = (_Float16*)(arena + (48ULL << 20));    // [262144,64]
    _Float16* A2  = (_Float16*)(arena);                    // [131072,160]
    _Float16* M2h = (_Float16*)(arena + (40ULL << 20));    // [131072,128]
    _Float16* O2h = (_Float16*)(arena + (72ULL << 20));    // [131072,256]

    // ---- pack all weights/biases (f32 -> padded transposed f16 / padded f32)
    for (int l = 0; l < 14; ++l) {
        int tot = L[l].Kpad * L[l].Npad;
        packwt_kernel<<<(tot + 255) / 256, 256, 0, stream>>>(
            P[L[l].wi], wp[l], L[l].K, L[l].N, L[l].Kpad, L[l].Npad);
        packb_kernel<<<1, L[l].Npad, 0, stream>>>(P[L[l].bi], bp[l], L[l].N, L[l].Npad);
    }

    // ---- SA1
    fps_kernel<<<1, 1024, 16384 * sizeof(float), stream>>>(pos, 16384, 8192, idx1);
    gather3_kernel<<<(8192 + 255) / 256, 256, 0, stream>>>(pos, idx1, pos1, 8192);
    knn_kernel<32><<<(8192 + 63) / 64, 64, 0, stream>>>(pos, 16384, pos1, 8192, nbr1);
    build_sa1_kernel<<<(262144 + 255) / 256, 256, 0, stream>>>(x, pos, pos1, nbr1, A1, 262144);
    lg<32, 1, 1>(A1, wp[0], bp[0], H1, 262144, 64, 64, 64, stream);
    lg<64, 0, 1>(H1, wp[1], bp[1], H2, 262144, 64, 64, 64, stream);
    maxpool_kernel<<<(8192 * 64 + 255) / 256, 256, 0, stream>>>(H2, P1h, 8192, 32, 64);
    lg<64, 1, 1>(P1h, wp[2], bp[2], G1h, 8192, 128, 128, 128, stream);
    lg<128, 0, 0>(G1h, wp[3], bp[3], x1f, 8192, 128, 128, 128, stream);

    // ---- SA2
    fps_kernel<<<1, 1024, 8192 * sizeof(float), stream>>>(pos1, 8192, 2048, idx2);
    gather3_kernel<<<(2048 + 255) / 256, 256, 0, stream>>>(pos1, idx2, pos2, 2048);
    knn_kernel<64><<<(2048 + 63) / 64, 64, 0, stream>>>(pos1, 8192, pos2, 2048, nbr2);
    build_sa2_kernel<<<(131072 + 255) / 256, 256, 0, stream>>>(x1f, pos1, pos2, nbr2, A2, 131072);
    lg<160, 1, 1>(A2, wp[4], bp[4], M2h, 131072, 128, 128, 128, stream);
    lg<128, 0, 1>(M2h, wp[5], bp[5], O2h, 131072, 256, 256, 256, stream);
    maxpool_kernel<<<(2048 * 256 + 255) / 256, 256, 0, stream>>>(O2h, P2h, 2048, 64, 256);
    lg<256, 1, 1>(P2h, wp[6], bp[6], G2h, 2048, 256, 256, 256, stream);
    lg<256, 0, 0>(G2h, wp[7], bp[7], x2f, 2048, 256, 256, 256, stream);

    // ---- FP2: interpolate x1 (pos1 -> pos2), concat with x2
    knn3w_kernel<<<(2048 + 63) / 64, 64, 0, stream>>>(pos1, 8192, pos2, 2048, idxA, wA);
    interp_kernel<<<(2048 * 128 + 255) / 256, 256, 0, stream>>>(x1f, 128, idxA, wA, xint2, 2048 * 128);
    build_fp_kernel<<<(2048 + 255) / 256, 256, 0, stream>>>(x2f, 256, xint2, 128, Afp2, 384, 2048);
    lg<384, 1, 1>(Afp2, wp[8], bp[8], T1h, 2048, 256, 256, 256, stream);
    lg<256, 0, 0>(T1h, wp[9], bp[9], xf2f, 2048, 256, 256, 256, stream);

    // ---- FP1: interpolate x (pos -> pos2), concat with xf2
    knn3w_kernel<<<(2048 + 63) / 64, 64, 0, stream>>>(pos, 16384, pos2, 2048, idxB, wB);
    interp_kernel<<<(2048 * 4 + 255) / 256, 256, 0, stream>>>(x, 4, idxB, wB, xint1, 2048 * 4);
    build_fp_kernel<<<(2048 + 255) / 256, 256, 0, stream>>>(xf2f, 256, xint1, 4, Afp1, 288, 2048);
    lg<288, 1, 1>(Afp1, wp[10], bp[10], Tb1h, 2048, 256, 256, 256, stream);
    lg<256, 0, 1>(Tb1h, wp[11], bp[11], xf1h, 2048, 256, 256, 256, stream);

    // ---- heads: sem [2048,3] then emb [2048,64] into d_out (f32)
    lg<256, 0, 0>(xf1h, wp[12], bp[12], (float*)d_out,            2048, 16, 3, 3, stream);
    lg<256, 0, 0>(xf1h, wp[13], bp[13], (float*)d_out + 2048 * 3, 2048, 64, 64, 64, stream);
}
